// PointTransformerV3_463856468483
// MI455X (gfx1250) — matrix operand order repaired and checked by reading, not compiled, not run on hardware
//
#include <hip/hip_runtime.h>
#include <vector>

// ---------------------------------------------------------------------------
// CDNA5 WMMA types
// ---------------------------------------------------------------------------
typedef __attribute__((ext_vector_type(16))) __bf16 v16bf;
typedef __attribute__((ext_vector_type(8)))  float  v8f;

union FragB {
  v16bf v;
  unsigned short h[16];
  unsigned int   u[8];
};

__device__ __forceinline__ unsigned short f2bf(float f) {
  unsigned u = __float_as_uint(f);
  u += 0x7fffu + ((u >> 16) & 1u);          // round-to-nearest-even
  return (unsigned short)(u >> 16);
}
__device__ __forceinline__ unsigned ld32u(const unsigned short* p) {
  return *(const unsigned*)p;               // callers guarantee 4B alignment
}
__device__ __forceinline__ v8f zero8() {
  v8f z = {0.f, 0.f, 0.f, 0.f, 0.f, 0.f, 0.f, 0.f};
  return z;
}
__device__ __forceinline__ float geluf(float x) {
  return 0.5f * x * (1.f + erff(x * 0.70710678118654752f));
}

// ---------------------------------------------------------------------------
// Elementwise / utility kernels
// ---------------------------------------------------------------------------
__global__ void k_fill_i32(int* p, int v, int n) {
  int i = blockIdx.x * blockDim.x + threadIdx.x;
  if (i < n) p[i] = v;
}
__global__ void k_f2bf(const float* s, unsigned short* d, int n) {
  int i = blockIdx.x * blockDim.x + threadIdx.x;
  if (i < n) d[i] = f2bf(s[i]);
}
__global__ void k_gelu_bf(const float* s, unsigned short* d, int n) {
  int i = blockIdx.x * blockDim.x + threadIdx.x;
  if (i < n) d[i] = f2bf(geluf(s[i]));
}
// concat(xyz, feats) into K=32-padded bf16 rows
__global__ void k_build_in8(const float* xyz, const float* feats, unsigned short* d, int T) {
  int i = blockIdx.x * blockDim.x + threadIdx.x;
  if (i >= T * 32) return;
  int row = i / 32, c = i % 32;
  float v = 0.f;
  if (c < 3) v = xyz[(size_t)row * 3 + c];
  else if (c < 8) v = feats[(size_t)row * 5 + (c - 3)];
  d[i] = f2bf(v);
}
__global__ void k_gather_rows(const float* src, float* dst, const int* order,
                              int Nl, int C, int total) {
  int i = blockIdx.x * blockDim.x + threadIdx.x;
  if (i >= total) return;
  int row = i / C, c = i % C;
  int b = row / Nl;
  dst[i] = src[((size_t)b * Nl + order[row]) * C + c];
}
__global__ void k_scatter_rows(const float* src, float* dst, const int* order,
                               int Nl, int C, int total) {
  int i = blockIdx.x * blockDim.x + threadIdx.x;
  if (i >= total) return;
  int row = i / C, c = i % C;
  int b = row / Nl;
  dst[((size_t)b * Nl + order[row]) * C + c] = src[i];
}
__global__ void k_expand_pad(const unsigned short* src, unsigned short* dst,
                             int Nl, int Np, int C, int total) {
  int i = blockIdx.x * blockDim.x + threadIdx.x;
  if (i >= total) return;
  int row = i / C, c = i % C;
  int b = row / Np, t = row % Np;
  dst[i] = (t < Nl) ? src[((size_t)b * Nl + t) * C + c] : (unsigned short)0;
}
__global__ void k_compact_pad(const unsigned short* src, unsigned short* dst,
                              int Nl, int Np, int C, int total) {
  int i = blockIdx.x * blockDim.x + threadIdx.x;
  if (i >= total) return;
  int row = i / C, c = i % C;
  int b = row / Nl, t = row % Nl;
  dst[i] = src[((size_t)b * Np + t) * C + c];
}
// concat( x_up[cluster[row]] (C1) , skip[row] (C0) ) -> bf16
__global__ void k_unpool_concat(const float* xup, const float* skip, const int* cluster,
                                int Nl, int M, int C1, int C0,
                                unsigned short* dst, int totalRows) {
  int Cc = C1 + C0;
  int i = blockIdx.x * blockDim.x + threadIdx.x;
  if (i >= totalRows * Cc) return;
  int row = i / Cc, c = i % Cc;
  int b = row / Nl;
  float v;
  if (c < C1) {
    int cl = cluster[row];
    v = xup[((size_t)b * M + cl) * C1 + c];
  } else {
    v = skip[(size_t)row * C0 + (c - C1)];
  }
  dst[i] = f2bf(v);
}

// ---------------------------------------------------------------------------
// Serialization / clustering kernels
// ---------------------------------------------------------------------------
__device__ __forceinline__ unsigned spread3(unsigned v) {
  v &= 0x3FFu;
  v = (v | (v << 16)) & 0x030000FFu;
  v = (v | (v << 8))  & 0x0300F00Fu;
  v = (v | (v << 4))  & 0x030C30C3u;
  v = (v | (v << 2))  & 0x09249249u;
  return v;
}
__global__ void k_coord_minmax(const float* xyz, float grid, int total, int Nl,
                               int* minb, int* maxb) {
  int i = blockIdx.x * blockDim.x + threadIdx.x;
  if (i >= total) return;
  int b = i / Nl;
  const float* p = xyz + (size_t)i * 3;
#pragma unroll
  for (int d = 0; d < 3; ++d) {
    int c = (int)floorf(p[d] / grid);
    atomicMin(&minb[b * 3 + d], c);
    atomicMax(&maxb[b * 3 + d], c);
  }
}
__global__ void k_morton_keys(const float* xyz, const int* minb, float grid,
                              int Nl, int np2, unsigned long long* keys, int total) {
  int i = blockIdx.x * blockDim.x + threadIdx.x;
  if (i >= total) return;
  int b = i / np2, t = i % np2;
  if (t >= Nl) { keys[i] = ~0ull; return; }
  const float* p = xyz + ((size_t)b * Nl + t) * 3;
  unsigned cx = (unsigned)((int)floorf(p[0] / grid) - minb[b * 3 + 0]);
  unsigned cy = (unsigned)((int)floorf(p[1] / grid) - minb[b * 3 + 1]);
  unsigned cz = (unsigned)((int)floorf(p[2] / grid) - minb[b * 3 + 2]);
  unsigned code = spread3(cx) | (spread3(cy) << 1) | (spread3(cz) << 2);
  keys[i] = ((unsigned long long)code << 32) | (unsigned)t;
}
__global__ void k_cluster_keys(const float* xyz, const int* minb, const int* maxb,
                               float grid, int Nl, int np2,
                               unsigned long long* keys, int total) {
  int i = blockIdx.x * blockDim.x + threadIdx.x;
  if (i >= total) return;
  int b = i / np2, t = i % np2;
  if (t >= Nl) { keys[i] = ~0ull; return; }
  const float* p = xyz + ((size_t)b * Nl + t) * 3;
  int c0 = (int)floorf(p[0] / grid) - minb[b * 3 + 0];
  int c1 = (int)floorf(p[1] / grid) - minb[b * 3 + 1];
  int c2 = (int)floorf(p[2] / grid) - minb[b * 3 + 2];
  int d1 = maxb[b * 3 + 1] - minb[b * 3 + 1] + 1;
  int d2 = maxb[b * 3 + 2] - minb[b * 3 + 2] + 1;
  unsigned key = (unsigned)(c0 * d1 * d2 + c1 * d2 + c2);
  keys[i] = ((unsigned long long)key << 32) | (unsigned)t;
}
__global__ void k_bitonic(unsigned long long* d, int n, int j, int k, int total) {
  int i = blockIdx.x * blockDim.x + threadIdx.x;
  if (i >= total) return;
  int t = i % n;
  unsigned long long* a = d + (size_t)(i / n) * n;
  int ixj = t ^ j;
  if (ixj > t) {
    unsigned long long x = a[t], y = a[ixj];
    bool up = ((t & k) == 0);
    if ((x > y) == up) { a[t] = y; a[ixj] = x; }
  }
}
__global__ void k_extract_order(const unsigned long long* keys, int np2, int Nl,
                                int* order, int total) {
  int i = blockIdx.x * blockDim.x + threadIdx.x;
  if (i >= total) return;
  int b = i / Nl, t = i % Nl;
  order[i] = (int)(keys[(size_t)b * np2 + t] & 0xffffffffu);
}
// per-batch dense rank of sorted keys + segment starts + inverse map
__global__ void k_rank_scan(const unsigned long long* keys, int np2, int n, int M,
                            int* cluster, int* segstart) {
  int b = blockIdx.x;
  const unsigned long long* kb = keys + (size_t)b * np2;
  int* clb = cluster + (size_t)b * n;
  int* ssb = segstart + (size_t)b * (M + 1);
  __shared__ int sh[256];
  __shared__ int carry;
  if (threadIdx.x == 0) carry = 0;
  __syncthreads();
  for (int c0 = 0; c0 < n; c0 += 256) {
    int j = c0 + (int)threadIdx.x;
    int flag = 0;
    unsigned long long kj = 0;
    if (j < n) {
      kj = kb[j];
      unsigned hi = (unsigned)(kj >> 32);
      flag = (j == 0) ? 1 : (hi != (unsigned)(kb[j - 1] >> 32));
    }
    sh[threadIdx.x] = flag;
    __syncthreads();
    for (int off = 1; off < 256; off <<= 1) {
      int v = sh[threadIdx.x];
      int add = (threadIdx.x >= (unsigned)off) ? sh[threadIdx.x - off] : 0;
      __syncthreads();
      sh[threadIdx.x] = v + add;
      __syncthreads();
    }
    int incl = sh[threadIdx.x] + carry;
    if (j < n) {
      int rank = incl - 1;
      int orig = (int)(kj & 0xffffffffu);
      clb[orig] = (rank < M) ? rank : (M - 1);
      if (flag && rank < M) ssb[rank] = j;
    }
    __syncthreads();
    if (threadIdx.x == 255) carry = incl;
    __syncthreads();
  }
}
// deterministic segment means (xyz + feats)
__global__ void k_seg_mean(const unsigned long long* keys, const int* segstart,
                           const float* xyz, const float* feats,
                           int np2, int Nl, int M, int C,
                           float* xyzp, float* featp) {
  int s = blockIdx.x % M, b = blockIdx.x / M;
  const unsigned long long* kb = keys + (size_t)b * np2;
  const int* ssb = segstart + (size_t)b * (M + 1);
  int start = ssb[s], end = ssb[s + 1];
  int cnt = end - start;
  float denom = (float)(cnt > 0 ? cnt : 1);
  for (int c = threadIdx.x; c < C + 3; c += blockDim.x) {
    float sum = 0.f;
    for (int j = start; j < end; ++j) {
      int row = (int)(kb[j] & 0xffffffffu);
      sum += (c < 3) ? xyz[((size_t)b * Nl + row) * 3 + c]
                     : feats[((size_t)b * Nl + row) * C + (c - 3)];
    }
    if (c < 3) xyzp[((size_t)b * M + s) * 3 + c] = sum / denom;
    else       featp[((size_t)b * M + s) * C + (c - 3)] = sum / denom;
  }
}

// ---------------------------------------------------------------------------
// LayerNorm (one wave per token)
// ---------------------------------------------------------------------------
__global__ void k_layernorm(const float* x, const float* g, const float* bb,
                            float* outf, unsigned short* outb, int T, int C) {
  int tok = (int)((blockIdx.x * blockDim.x + threadIdx.x) >> 5);
  if (tok >= T) return;
  int l = threadIdx.x & 31;
  const float* xr = x + (size_t)tok * C;
  float s = 0.f, s2 = 0.f;
  for (int c = l; c < C; c += 32) { float v = xr[c]; s += v; s2 += v * v; }
  for (int m = 16; m; m >>= 1) { s += __shfl_xor(s, m, 32); s2 += __shfl_xor(s2, m, 32); }
  float mean = s / C;
  float var = s2 / C - mean * mean;
  var = var < 0.f ? 0.f : var;
  float inv = rsqrtf(var + 1e-5f);
  for (int c = l; c < C; c += 32) {
    float v = (xr[c] - mean) * inv * g[c] + bb[c];
    if (outf) outf[(size_t)tok * C + c] = v;
    if (outb) outb[(size_t)tok * C + c] = f2bf(v);
  }
}

// ---------------------------------------------------------------------------
// WMMA GEMM: out[M,N] = A[M,K](bf16) * W[N,K]^T(bf16) + bias, epilogues
// wave -> 16 rows x 64 cols, block(256) -> 128 rows x 64 cols
// ---------------------------------------------------------------------------
template <bool GELU, bool RES, bool OUTF, bool OUTB>
__global__ __launch_bounds__(256) void k_gemm(
    const unsigned short* __restrict__ A, int lda,
    const unsigned short* __restrict__ W, int ldw,
    const float* __restrict__ bias, const float* __restrict__ res,
    float* __restrict__ outf, unsigned short* __restrict__ outb,
    int ldo, int M, int N, int K) {
  int lane = threadIdx.x & 31, wave = (int)(threadIdx.x >> 5);
  int row0 = blockIdx.x * 128 + wave * 16;
  int col0 = blockIdx.y * 64;
  if (row0 >= M) return;
  int half = lane >> 4, mn = lane & 15;
  v8f acc[4];
#pragma unroll
  for (int j = 0; j < 4; ++j) acc[j] = zero8();

  for (int k0 = 0; k0 < K; k0 += 32) {
    FragB af;
    int arow = row0 + mn;
#pragma unroll
    for (int p = 0; p < 8; ++p) {
      int kl = 2 * p + (p >= 4 ? 8 : 0) + 8 * half;
      int k = k0 + kl;
      af.u[p] = (k < K) ? ld32u(A + (size_t)arow * lda + k) : 0u;
    }
#pragma unroll
    for (int j = 0; j < 4; ++j) {
      FragB bf;
      int col = col0 + 16 * j + mn;
      int colc = (col < N) ? col : (N - 1);
#pragma unroll
      for (int p = 0; p < 8; ++p) {
        int k = k0 + 2 * p + (p >= 4 ? 8 : 0) + 8 * half;
        bf.u[p] = (k < K) ? ld32u(W + (size_t)colc * ldw + k) : 0u;
      }
      acc[j] = __builtin_amdgcn_wmma_f32_16x16x32_bf16(
          false, af.v, false, bf.v, (short)0, acc[j], false, false);
    }
  }
#pragma unroll
  for (int j = 0; j < 4; ++j) {
    int col = col0 + 16 * j + mn;
    if (col >= N) continue;
    float bv = bias ? bias[col] : 0.f;
#pragma unroll
    for (int r = 0; r < 8; ++r) {
      int row = row0 + 8 * half + r;
      if (row >= M) continue;
      float v = acc[j][r] + bv;
      if (GELU) v = geluf(v);
      if (RES)  v += res[(size_t)row * ldo + col];
      if (OUTF) outf[(size_t)row * ldo + col] = v;
      if (OUTB) outb[(size_t)row * ldo + col] = f2bf(v);
    }
  }
}

// ---------------------------------------------------------------------------
// Fused windowed attention: one WG per (window, head). WIN=256, hd=16.
// block = 128 threads (4 waves), each wave does 4 row-tiles of 16 queries.
// ---------------------------------------------------------------------------
__global__ __launch_bounds__(128) void k_attn(const unsigned short* __restrict__ qkv,
                                              unsigned short* __restrict__ outb, int C) {
  int win = blockIdx.x, h = blockIdx.y;
  int tid = threadIdx.x, lane = tid & 31, wave = tid >> 5;
  int base = win * 256;
  int ld = 3 * C;
  __shared__ __align__(16) unsigned short kt[256 * 16];  // [token][d]
  __shared__ __align__(16) unsigned short vt[16 * 256];  // [d][token] (transposed)
  __shared__ __align__(16) unsigned short pt[4][16 * 256];

  for (int t = tid; t < 256; t += 128) {
    const unsigned short* kr = qkv + (size_t)(base + t) * ld + C + h * 16;
    const unsigned short* vr = qkv + (size_t)(base + t) * ld + 2 * C + h * 16;
#pragma unroll
    for (int d = 0; d < 16; ++d) {
      kt[t * 16 + d] = kr[d];
      vt[d * 256 + t] = vr[d];
    }
  }
  __syncthreads();

  int half = lane >> 4, mn = lane & 15;
  for (int rt = 0; rt < 4; ++rt) {
    int row0 = (wave * 4 + rt) * 16;
    // Q A-frag (K=16 real, upper 16 zero)
    FragB aq;
#pragma unroll
    for (int p = 0; p < 8; ++p) {
      int kl = 2 * p + (p >= 4 ? 8 : 0) + 8 * half;
      aq.u[p] = (kl < 16)
                    ? ld32u(qkv + (size_t)(base + row0 + mn) * ld + h * 16 + kl)
                    : 0u;
    }
    // S = Q * K^T : 16 accumulators of 16x16
    v8f acc[16];
#pragma unroll
    for (int j = 0; j < 16; ++j) {
      FragB bf;
#pragma unroll
      for (int p = 0; p < 8; ++p) {
        int k = 2 * p + (p >= 4 ? 8 : 0) + 8 * half;
        bf.u[p] = (k < 16) ? ld32u(kt + (16 * j + mn) * 16 + k) : 0u;
      }
      acc[j] = __builtin_amdgcn_wmma_f32_16x16x32_bf16(
          false, aq.v, false, bf.v, (short)0, zero8(), false, false);
    }
    // softmax rows (scale = hd^-0.5 = 0.25)
    float inv[8];
#pragma unroll
    for (int r = 0; r < 8; ++r) {
      float mx = -1e30f;
#pragma unroll
      for (int j = 0; j < 16; ++j) mx = fmaxf(mx, acc[j][r]);
      for (int d = 1; d < 16; d <<= 1) mx = fmaxf(mx, __shfl_xor(mx, d, 32));
      float sum = 0.f;
#pragma unroll
      for (int j = 0; j < 16; ++j) {
        float e = __expf((acc[j][r] - mx) * 0.25f);
        acc[j][r] = e;
        sum += e;
      }
      for (int d = 1; d < 16; d <<= 1) sum += __shfl_xor(sum, d, 32);
      inv[r] = 1.f / sum;
    }
    // write P (bf16, row-major 16x256) to this wave's LDS tile
#pragma unroll
    for (int j = 0; j < 16; ++j)
#pragma unroll
      for (int r = 0; r < 8; ++r) {
        int m = 8 * half + r, n = 16 * j + mn;
        pt[wave][m * 256 + n] = f2bf(acc[j][r] * inv[r]);
      }
    // O = P * V : accumulate 8 K-chunks of 32
    v8f o = zero8();
#pragma unroll
    for (int kc = 0; kc < 8; ++kc) {
      FragB ap, bv;
#pragma unroll
      for (int p = 0; p < 8; ++p) {
        int kl = 2 * p + (p >= 4 ? 8 : 0) + 8 * half;
        ap.u[p] = ld32u(&pt[wave][mn * 256 + kc * 32 + kl]);
      }
#pragma unroll
      for (int p = 0; p < 8; ++p) {
        int k = kc * 32 + 2 * p + (p >= 4 ? 8 : 0) + 8 * half;
        bv.u[p] = ld32u(vt + mn * 256 + k);
      }
      o = __builtin_amdgcn_wmma_f32_16x16x32_bf16(
          false, ap.v, false, bv.v, (short)0, o, false, false);
    }
#pragma unroll
    for (int r = 0; r < 8; ++r) {
      int row = base + row0 + 8 * half + r;
      outb[(size_t)row * C + h * 16 + mn] = f2bf(o[r]);
    }
  }
}

// ---------------------------------------------------------------------------
// Host orchestration
// ---------------------------------------------------------------------------
struct LinP { const float* w; const float* b; unsigned short* wbf; int din, dout; };
struct LnP  { const float* g; const float* b; };
struct BlkP { LnP ln1; LinP qkv, proj; LnP ln2; LinP fc1, fc2; };

extern "C" void kernel_launch(void* const* d_in, const int* in_sizes, int n_in,
                              void* d_out, int out_size, void* d_ws, size_t ws_size,
                              hipStream_t stream) {
  (void)in_sizes; (void)n_in; (void)out_size;
  const int B = 4;
  const int NL[4]  = {16384, 2560, 512, 128};
  const int NP2[4] = {16384, 4096, 512, 128};   // pow2 sort sizes per level
  const int CD[4]  = {48, 96, 192, 384};
  const int HH[4]  = {3, 6, 12, 24};
  const int DEP[4] = {2, 2, 6, 2};
  const int MS[3]  = {2560, 512, 128};
  const float GR[3] = {0.08f, 0.16f, 0.32f};
  const float SER = 0.04f;

  // ---- parse inputs (setup_inputs() insertion order, recursive) ----
  int idx = 0;
  auto getf = [&]() { return (const float*)d_in[idx++]; };
  const float* xyz0  = getf();
  const float* feats = getf();

  std::vector<LinP*> allLin;
  auto getLin = [&](int di, int dout_) {
    LinP L; L.w = getf(); L.b = getf(); L.wbf = nullptr; L.din = di; L.dout = dout_;
    return L;
  };
  auto getLn = [&]() { LnP L; L.g = getf(); L.b = getf(); return L; };

  LinP ipLin1 = getLin(8, 48);   LnP ipLn1 = getLn();
  LinP ipLin2 = getLin(48, 48);  LnP ipLn2 = getLn();

  BlkP enc[4][6];
  auto getBlk = [&](int d) {
    BlkP bp;
    bp.ln1 = getLn();
    bp.qkv = getLin(d, 3 * d);
    bp.proj = getLin(d, d);
    bp.ln2 = getLn();
    bp.fc1 = getLin(d, 4 * d);
    bp.fc2 = getLin(4 * d, d);
    return bp;
  };
  for (int i = 0; i < 4; ++i)
    for (int dd = 0; dd < DEP[i]; ++dd) enc[i][dd] = getBlk(CD[i]);

  LinP poolLin[3]; LnP poolLn[3];
  for (int i = 0; i < 3; ++i) { poolLin[i] = getLin(CD[i], CD[i + 1]); poolLn[i] = getLn(); }
  LinP upLin[3]; LnP upLn[3];
  for (int j = 0; j < 3; ++j) {
    int i = 2 - j;
    upLin[j] = getLin(CD[i + 1] + CD[i], CD[i]); upLn[j] = getLn();
  }
  BlkP dec[3];
  for (int j = 0; j < 3; ++j) dec[j] = getBlk(CD[2 - j]);
  LinP headL = getLin(48, 8);

  // collect all linears for bf16 conversion
  auto collect = [&](BlkP& bp) {
    allLin.push_back(&bp.qkv); allLin.push_back(&bp.proj);
    allLin.push_back(&bp.fc1); allLin.push_back(&bp.fc2);
  };
  allLin.push_back(&ipLin1); allLin.push_back(&ipLin2);
  for (int i = 0; i < 4; ++i) for (int dd = 0; dd < DEP[i]; ++dd) collect(enc[i][dd]);
  for (int i = 0; i < 3; ++i) allLin.push_back(&poolLin[i]);
  for (int j = 0; j < 3; ++j) allLin.push_back(&upLin[j]);
  for (int j = 0; j < 3; ++j) collect(dec[j]);
  allLin.push_back(&headL);

  // ---- workspace bump allocator ----
  char* wsbase = (char*)d_ws;
  size_t wsoff = 0;
  auto alloc = [&](size_t bytes) -> void* {
    void* p = wsbase + wsoff;
    wsoff = (wsoff + bytes + 255) & ~(size_t)255;
    return p;
  };
  for (auto* L : allLin) L->wbf = (unsigned short*)alloc((size_t)L->din * L->dout * 2);

  float* xyz_l[4];
  xyz_l[0] = nullptr;  // input
  for (int i = 1; i < 4; ++i) xyz_l[i] = (float*)alloc((size_t)B * NL[i] * 3 * 4);
  float* x_enc[4]; float* x_dec[3];
  for (int i = 0; i < 4; ++i) x_enc[i] = (float*)alloc((size_t)B * NL[i] * CD[i] * 4);
  for (int i = 0; i < 3; ++i) x_dec[i] = (float*)alloc((size_t)B * NL[i] * CD[i] * 4);
  int* order_l[4]; int* cluster_l[3];
  for (int i = 0; i < 4; ++i) order_l[i] = (int*)alloc((size_t)B * NL[i] * 4);
  for (int i = 0; i < 3; ++i) cluster_l[i] = (int*)alloc((size_t)B * NL[i] * 4);
  unsigned long long* keys = (unsigned long long*)alloc((size_t)B * 16384 * 8);
  int* minb = (int*)alloc(B * 3 * 4);
  int* maxb = (int*)alloc(B * 3 * 4);
  int* segstart = (int*)alloc((size_t)B * (2560 + 1) * 4);
  float* featp = (float*)alloc((size_t)B * 2560 * 48 * 4);
  unsigned short* featp_bf = (unsigned short*)alloc((size_t)B * 2560 * 48 * 2);
  float* x_ord = (float*)alloc((size_t)65536 * 48 * 4);
  float* tmpf  = (float*)alloc((size_t)65536 * 48 * 4);
  unsigned short* xln_bf  = (unsigned short*)alloc((size_t)65536 * 48 * 2);
  unsigned short* big_bf  = (unsigned short*)alloc((size_t)65536 * 192 * 2);  // qkv/h/concat
  unsigned short* attn_bf = (unsigned short*)alloc((size_t)65536 * 48 * 2);
  unsigned short* xpad_bf = (unsigned short*)alloc((size_t)1024 * 384 * 2);
  unsigned short* apad_bf = (unsigned short*)alloc((size_t)1024 * 384 * 2);
  unsigned short* in8_bf  = (unsigned short*)alloc((size_t)65536 * 32 * 2);
  if (wsoff > ws_size) return;  // not enough scratch; nothing sane possible

  auto cdiv = [](int a, int b) { return (a + b - 1) / b; };

  // weight conversions (deterministic, every call)
  for (auto* L : allLin) {
    int n = L->din * L->dout;
    k_f2bf<<<cdiv(n, 256), 256, 0, stream>>>(L->w, L->wbf, n);
  }

  // ---- helpers ----
  auto gemm = [&](const unsigned short* A, int lda, const LinP& L,
                  const float* res, float* outf, unsigned short* outb,
                  int ldo, int M, int N, int K, bool gelu) {
    dim3 g(cdiv(M, 128), cdiv(N, 64));
    if (gelu)
      k_gemm<true, false, false, true><<<g, 256, 0, stream>>>(A, lda, L.wbf, L.din, L.b,
                                                              nullptr, nullptr, outb, ldo, M, N, K);
    else if (res)
      k_gemm<false, true, true, false><<<g, 256, 0, stream>>>(A, lda, L.wbf, L.din, L.b,
                                                              res, outf, nullptr, ldo, M, N, K);
    else if (outb)
      k_gemm<false, false, false, true><<<g, 256, 0, stream>>>(A, lda, L.wbf, L.din, L.b,
                                                               nullptr, nullptr, outb, ldo, M, N, K);
    else
      k_gemm<false, false, true, false><<<g, 256, 0, stream>>>(A, lda, L.wbf, L.din, L.b,
                                                               nullptr, outf, nullptr, ldo, M, N, K);
  };
  auto lnrun = [&](const float* x, const LnP& L, float* outf, unsigned short* outb,
                   int T, int C) {
    k_layernorm<<<cdiv(T * 32, 256), 256, 0, stream>>>(x, L.g, L.b, outf, outb, T, C);
  };
  auto sortKeys = [&](int np2) {
    int total = B * np2, blocks = cdiv(total, 256);
    for (int k = 2; k <= np2; k <<= 1)
      for (int j = k >> 1; j > 0; j >>= 1)
        k_bitonic<<<blocks, 256, 0, stream>>>(keys, np2, j, k, total);
  };
  auto makeOrder = [&](const float* xyzl, int Nl, int np2, int* order) {
    k_fill_i32<<<cdiv(B * 3, 64), 64, 0, stream>>>(minb, 0x7fffffff, B * 3);
    k_fill_i32<<<cdiv(B * 3, 64), 64, 0, stream>>>(maxb, (int)0x80000000, B * 3);
    k_coord_minmax<<<cdiv(B * Nl, 256), 256, 0, stream>>>(xyzl, SER, B * Nl, Nl, minb, maxb);
    k_morton_keys<<<cdiv(B * np2, 256), 256, 0, stream>>>(xyzl, minb, SER, Nl, np2, keys, B * np2);
    sortKeys(np2);
    k_extract_order<<<cdiv(B * Nl, 256), 256, 0, stream>>>(keys, np2, Nl, order, B * Nl);
  };
  auto runBlock = [&](BlkP& bp, float* xord, int Tl, int NlPB, int NpPB, int C, int H) {
    int Tp = B * NpPB;
    lnrun(xord, bp.ln1, nullptr, xln_bf, Tl, C);
    const unsigned short* qin = xln_bf;
    if (Tp != Tl) {
      k_expand_pad<<<cdiv(Tp * C, 256), 256, 0, stream>>>(xln_bf, xpad_bf, NlPB, NpPB, C, Tp * C);
      qin = xpad_bf;
    }
    gemm(qin, C, bp.qkv, nullptr, nullptr, big_bf, 3 * C, Tp, 3 * C, C, false);
    unsigned short* aout = (Tp != Tl) ? apad_bf : attn_bf;
    dim3 ag(Tp / 256, H);
    k_attn<<<ag, 128, 0, stream>>>(big_bf, aout, C);
    if (Tp != Tl)
      k_compact_pad<<<cdiv(Tl * C, 256), 256, 0, stream>>>(apad_bf, attn_bf, NlPB, NpPB, C, Tl * C);
    gemm(attn_bf, C, bp.proj, xord, xord, nullptr, C, Tl, C, C, false);
    lnrun(xord, bp.ln2, nullptr, xln_bf, Tl, C);
    gemm(xln_bf, C, bp.fc1, nullptr, nullptr, big_bf, 4 * C, Tl, 4 * C, C, true);
    gemm(big_bf, 4 * C, bp.fc2, xord, xord, nullptr, C, Tl, C, 4 * C, false);
  };

  // ---- input projection ----
  const int T0 = B * NL[0];
  k_build_in8<<<cdiv(T0 * 32, 256), 256, 0, stream>>>(xyz0, feats, in8_bf, T0);
  gemm(in8_bf, 32, ipLin1, nullptr, tmpf, nullptr, 48, T0, 48, 8, false);
  lnrun(tmpf, ipLn1, tmpf, nullptr, T0, 48);            // in-place LN
  k_gelu_bf<<<cdiv(T0 * 48, 256), 256, 0, stream>>>(tmpf, xln_bf, T0 * 48);
  gemm(xln_bf, 48, ipLin2, nullptr, tmpf, nullptr, 48, T0, 48, 48, false);
  lnrun(tmpf, ipLn2, x_enc[0], nullptr, T0, 48);

  // ---- encoder ----
  const float* xyzl_ptr[4] = {xyz0, xyz_l[1], xyz_l[2], xyz_l[3]};
  for (int i = 0; i < 4; ++i) {
    int Tl = B * NL[i], C = CD[i];
    makeOrder(xyzl_ptr[i], NL[i], NP2[i], order_l[i]);
    k_gather_rows<<<cdiv(Tl * C, 256), 256, 0, stream>>>(x_enc[i], x_ord, order_l[i], NL[i], C, Tl * C);
    for (int dd = 0; dd < DEP[i]; ++dd)
      runBlock(enc[i][dd], x_ord, Tl, NL[i], (i == 3 ? 256 : NL[i]), C, HH[i]);
    k_scatter_rows<<<cdiv(Tl * C, 256), 256, 0, stream>>>(x_ord, x_enc[i], order_l[i], NL[i], C, Tl * C);
    if (i < 3) {
      int M = MS[i], np2 = NP2[i];
      k_fill_i32<<<cdiv(B * 3, 64), 64, 0, stream>>>(minb, 0x7fffffff, B * 3);
      k_fill_i32<<<cdiv(B * 3, 64), 64, 0, stream>>>(maxb, (int)0x80000000, B * 3);
      k_coord_minmax<<<cdiv(Tl, 256), 256, 0, stream>>>(xyzl_ptr[i], GR[i], Tl, NL[i], minb, maxb);
      k_cluster_keys<<<cdiv(B * np2, 256), 256, 0, stream>>>(xyzl_ptr[i], minb, maxb, GR[i],
                                                             NL[i], np2, keys, B * np2);
      sortKeys(np2);
      k_fill_i32<<<cdiv(B * (M + 1), 256), 256, 0, stream>>>(segstart, NL[i], B * (M + 1));
      k_rank_scan<<<B, 256, 0, stream>>>(keys, np2, NL[i], M, cluster_l[i], segstart);
      k_seg_mean<<<B * M, 64, 0, stream>>>(keys, segstart, xyzl_ptr[i], x_enc[i],
                                           np2, NL[i], M, C, xyz_l[i + 1], featp);
      int pn = B * M * C;
      k_f2bf<<<cdiv(pn, 256), 256, 0, stream>>>(featp, featp_bf, pn);
      gemm(featp_bf, C, poolLin[i], nullptr, tmpf, nullptr, CD[i + 1], B * M, CD[i + 1], C, false);
      lnrun(tmpf, poolLn[i], x_enc[i + 1], nullptr, B * M, CD[i + 1]);
    }
  }

  // ---- decoder ----
  float* xcur = x_enc[3];
  for (int j = 0; j < 3; ++j) {
    int i = 2 - j;
    int C1 = CD[i + 1], C0 = CD[i], Tl = B * NL[i], M = NL[i + 1];
    k_unpool_concat<<<cdiv(Tl * (C1 + C0), 256), 256, 0, stream>>>(
        xcur, x_enc[i], cluster_l[i], NL[i], M, C1, C0, big_bf, Tl);
    gemm(big_bf, C1 + C0, upLin[j], nullptr, tmpf, nullptr, C0, Tl, C0, C1 + C0, false);
    lnrun(tmpf, upLn[j], x_dec[i], nullptr, Tl, C0);
    k_gather_rows<<<cdiv(Tl * C0, 256), 256, 0, stream>>>(x_dec[i], x_ord, order_l[i], NL[i], C0, Tl * C0);
    runBlock(dec[j], x_ord, Tl, NL[i], NL[i], C0, HH[i]);
    k_scatter_rows<<<cdiv(Tl * C0, 256), 256, 0, stream>>>(x_ord, x_dec[i], order_l[i], NL[i], C0, Tl * C0);
    xcur = x_dec[i];
  }

  // ---- head ----
  k_f2bf<<<cdiv(T0 * 48, 256), 256, 0, stream>>>(x_dec[0], xln_bf, T0 * 48);
  gemm(xln_bf, 48, headL, nullptr, (float*)d_out, nullptr, 8, T0, 8, 48, false);
}
